// MultiScaleKNNGraphAttention_9079560864225
// MI455X (gfx1250) — matrix operand order repaired and checked by reading, not compiled, not run on hardware
//
#include <hip/hip_runtime.h>

// ---------------------------------------------------------------------------
// MultiScaleKNNGraphAttention for MI455X (gfx1250, wave32, WMMA + async LDS)
// ---------------------------------------------------------------------------

typedef __attribute__((ext_vector_type(16))) __bf16 v16bf;
typedef __attribute__((ext_vector_type(8)))  float  v8f;

#define WAVE 32

__device__ __forceinline__ unsigned short f32_to_bf16_rne(float f) {
  unsigned u = __builtin_bit_cast(unsigned, f);
  u += 0x7FFFu + ((u >> 16) & 1u);          // round-to-nearest-even
  return (unsigned short)(u >> 16);
}

__device__ __forceinline__ float bf16_bits_to_f32(unsigned hi16) {
  return __builtin_bit_cast(float, hi16);
}

__device__ __forceinline__ float wave_sum(float v) {
#pragma unroll
  for (int m = 16; m >= 1; m >>= 1) v += __shfl_xor(v, m, WAVE);
  return v;
}

// one 16-byte lane-wise async copy: global -> LDS, tracked by ASYNCcnt
__device__ __forceinline__ void async_cp16(unsigned lds_off, const void* gaddr) {
  asm volatile("global_load_async_to_lds_b128 %0, %1, off"
               :: "v"(lds_off), "v"((unsigned long long)(size_t)gaddr)
               : "memory");
}

// ---------------------------------------------------------------------------
// bulk f32 -> bf16 conversion (vectorized, grid-stride free: exact sizing)
// ---------------------------------------------------------------------------
__global__ __launch_bounds__(256)
void f32_to_bf16_kernel(const float* __restrict__ src,
                        unsigned short* __restrict__ dst, const long n4)
{
  const long i = (long)blockIdx.x * 256 + threadIdx.x;
  if (i >= n4) return;
  const float4 v = ((const float4*)src)[i];
  ushort4 o;
  o.x = f32_to_bf16_rne(v.x); o.y = f32_to_bf16_rne(v.y);
  o.z = f32_to_bf16_rne(v.z); o.w = f32_to_bf16_rne(v.w);
  ((ushort4*)dst)[i] = o;
}

// ---------------------------------------------------------------------------
// Y[M x Nout] = act( A[M x Kdim](bf16) @ Bw[Nout x Kdim](bf16)^T + bias )
// Block: 256 threads (8 waves). Block tile 64x128, K-step 32, double-buffered
// LDS staged with GLOBAL_LOAD_ASYNC_TO_LDS_B128 (3 x 16B copies per thread
// per stage). Wave (w&3) -> 16-row group, (w>>2) -> 64-col group.
// flags: bit0 = relu, bit1 = bf16 output.
// Requires M%64==0, Nout%128==0, Kdim%32==0 (holds: 40000/64, 256|128, 256).
// ---------------------------------------------------------------------------
#define TM 64
#define TN 128
#define TK 32
#define LDSP 40   // padded LDS row stride (ushorts) -> conflict-free frag reads

__global__ __launch_bounds__(256)
void gemm_bf16_wmma(const unsigned short* __restrict__ A,
                    const unsigned short* __restrict__ Bw,
                    const float* __restrict__ bias,
                    void* __restrict__ Y,
                    const int Kdim, const int Nout, const int flags)
{
  __shared__ unsigned short As[2][TM][LDSP];
  __shared__ unsigned short Bs[2][TN][LDSP];

  const int tid  = threadIdx.x;
  const int lane = tid & 31;
  const int wave = tid >> 5;
  const long m0  = (long)blockIdx.x * TM;
  const int  n0  = blockIdx.y * TN;
  const int  wm  = (wave & 3) * 16;   // wave row offset in tile
  const int  wn  = (wave >> 2) * 64;  // wave col offset in tile

  // LDS byte offsets (generic-pointer low 32 bits == LDS byte address)
  const unsigned asBase = (unsigned)(size_t)&As[0][0][0];
  const unsigned bsBase = (unsigned)(size_t)&Bs[0][0][0];
  const unsigned aBuf   = TM * LDSP * 2u;  // bytes per A buffer
  const unsigned bBuf   = TN * LDSP * 2u;

  // copy assignment: 16B chunks of 8 bf16, K-contiguous
  const int ar  = tid >> 2;            // A: row 0..63
  const int ac8 = (tid & 3) * 8;       // A: k-chunk 0/8/16/24
  const int br  = tid >> 2;            // B: rows br and br+64
  const int bc8 = (tid & 3) * 8;

  auto stage = [&](int buf, int kk) {
    async_cp16(asBase + buf * aBuf + (unsigned)(ar * LDSP + ac8) * 2u,
               A + (m0 + ar) * (long)Kdim + kk + ac8);
    async_cp16(bsBase + buf * bBuf + (unsigned)(br * LDSP + bc8) * 2u,
               Bw + (long)(n0 + br) * Kdim + kk + bc8);
    async_cp16(bsBase + buf * bBuf + (unsigned)((br + 64) * LDSP + bc8) * 2u,
               Bw + (long)(n0 + br + 64) * Kdim + kk + bc8);
  };

  v8f acc[4] = {};
  stage(0, 0);
  int buf = 0;
  for (int kk = 0; kk < Kdim; kk += TK) {
    if (kk + TK < Kdim) {
      stage(buf ^ 1, kk + TK);                     // prefetch next tile
      asm volatile("s_wait_asynccnt 0x3" ::: "memory");  // current tile landed
    } else {
      asm volatile("s_wait_asynccnt 0x0" ::: "memory");
    }
    __syncthreads();

    // A fragment (16x32 bf16): lane<16 -> K 0..15; lane>=16 -> K 16..31
    v16bf a;
    {
      const unsigned short* ap = &As[buf][wm + (lane & 15)][(lane < 16) ? 0 : 16];
#pragma unroll
      for (int j = 0; j < 8; ++j) a[j]     = __builtin_bit_cast(__bf16, ap[j]);
#pragma unroll
      for (int j = 0; j < 8; ++j) a[8 + j] = __builtin_bit_cast(__bf16, ap[8 + j]);
    }
    // B fragments (32x16 bf16): lane<16 -> K 0..15; lane>=16 -> K 16..31
    const int kbB = (lane < 16) ? 0 : 16;
#pragma unroll
    for (int t = 0; t < 4; ++t) {
      const unsigned short* bp = &Bs[buf][wn + t * 16 + (lane & 15)][kbB];
      v16bf b;
#pragma unroll
      for (int j = 0; j < 16; ++j) b[j] = __builtin_bit_cast(__bf16, bp[j]);
      acc[t] = __builtin_amdgcn_wmma_f32_16x16x32_bf16(
          false, a, false, b, (short)0, acc[t], false, false);
    }
    __syncthreads();   // everyone done reading buf before it is re-staged
    buf ^= 1;
  }

  // C/D layout: VGPR j -> row j (lanes 0-15) / row 8+j (lanes 16-31), col = lane%16
  const int rowh = (lane < 16) ? 0 : 8;
  const bool relu = (flags & 1) != 0;
  const bool obf  = (flags & 2) != 0;
#pragma unroll
  for (int t = 0; t < 4; ++t) {
    const int col = n0 + wn + t * 16 + (lane & 15);
    const float bv = bias[col];
#pragma unroll
    for (int j = 0; j < 8; ++j) {
      float v = acc[t][j] + bv;
      if (relu) v = fmaxf(v, 0.f);
      const long row = m0 + wm + rowh + j;
      if (obf) ((unsigned short*)Y)[row * Nout + col] = f32_to_bf16_rne(v);
      else     ((float*)Y)[row * Nout + col] = v;
    }
  }
}

// ---------------------------------------------------------------------------
// alpha = softmax(h @ Wg2^T + bg2) ; one wave32 per point, lane owns 4 of 128
// ---------------------------------------------------------------------------
__global__ __launch_bounds__(256)
void gate_alpha_kernel(const float* __restrict__ h, const float* __restrict__ Wg2,
                       const float* __restrict__ bg2, float* __restrict__ alpha,
                       const int Mtot)
{
  const int gw = blockIdx.x * 8 + (threadIdx.x >> 5);
  if (gw >= Mtot) return;
  const int lane = threadIdx.x & 31;
  const float4 hv = *(const float4*)(h + (long)gw * 128 + lane * 4);
  float lg[3];
#pragma unroll
  for (int i = 0; i < 3; ++i) {
    const float4 wv = *(const float4*)(Wg2 + i * 128 + lane * 4);
    const float p = hv.x * wv.x + hv.y * wv.y + hv.z * wv.z + hv.w * wv.w;
    lg[i] = wave_sum(p) + bg2[i];
  }
  const float mx = fmaxf(lg[0], fmaxf(lg[1], lg[2]));
  const float e0 = __expf(lg[0] - mx), e1 = __expf(lg[1] - mx), e2 = __expf(lg[2] - mx);
  const float inv = 1.f / (e0 + e1 + e2);
  if (lane == 0) {
    alpha[gw * 3 + 0] = e0 * inv;
    alpha[gw * 3 + 1] = e1 * inv;
    alpha[gw * 3 + 2] = e2 * inv;
  }
}

// ---------------------------------------------------------------------------
// Per point (one wave32, lane owns 8 of 256 channels):
//   gather K=16 bf16 neighbors (L2-resident), s = Q.K/16, softmax, o = a.V,
//   O = LN(o + Q), out += alpha_i * O
// ---------------------------------------------------------------------------
__device__ __forceinline__ void load_bf16x8(const unsigned short* p, float f[8]) {
  const uint4 u = *(const uint4*)p;
  f[0] = bf16_bits_to_f32(u.x << 16); f[1] = bf16_bits_to_f32(u.x & 0xffff0000u);
  f[2] = bf16_bits_to_f32(u.y << 16); f[3] = bf16_bits_to_f32(u.y & 0xffff0000u);
  f[4] = bf16_bits_to_f32(u.z << 16); f[5] = bf16_bits_to_f32(u.z & 0xffff0000u);
  f[6] = bf16_bits_to_f32(u.w << 16); f[7] = bf16_bits_to_f32(u.w & 0xffff0000u);
}

__global__ __launch_bounds__(256)
void knn_attn_kernel(const float* __restrict__ Qb,
                     const unsigned short* __restrict__ Kb,
                     const unsigned short* __restrict__ Vb,
                     const int* __restrict__ knn,
                     const float* __restrict__ lng, const float* __restrict__ lnb,
                     const float* __restrict__ alpha, const int iblk,
                     float* __restrict__ out, const int Npts)
{
  const int gw = blockIdx.x * 8 + (threadIdx.x >> 5);   // point in [0, B*N)
  if (gw >= 2 * Npts) return;
  const int lane = threadIdx.x & 31;
  const int b = gw / Npts;
  const int n = gw - b * Npts;
  const int c0 = lane * 8;

  const float* qp = Qb + (long)gw * 256 + c0;
  const float4 q0 = *(const float4*)qp;
  const float4 q1 = *(const float4*)(qp + 4);
  const float q[8] = {q0.x, q0.y, q0.z, q0.w, q1.x, q1.y, q1.z, q1.w};

  int idx[16];
#pragma unroll
  for (int j = 0; j < 16; ++j) idx[j] = knn[n * 16 + j];

  const float scale = 0.0625f;   // 1/sqrt(256)
  float s[16];
#pragma unroll
  for (int j = 0; j < 16; ++j) {
    float kf[8];
    load_bf16x8(Kb + ((long)b * Npts + idx[j]) * 256 + c0, kf);
    float p = 0.f;
#pragma unroll
    for (int c = 0; c < 8; ++c) p += q[c] * kf[c];
    s[j] = wave_sum(p) * scale;
  }
  float mx = s[0];
#pragma unroll
  for (int j = 1; j < 16; ++j) mx = fmaxf(mx, s[j]);
  float den = 0.f;
#pragma unroll
  for (int j = 0; j < 16; ++j) { s[j] = __expf(s[j] - mx); den += s[j]; }
  const float inv = 1.f / den;

  float o[8] = {0.f, 0.f, 0.f, 0.f, 0.f, 0.f, 0.f, 0.f};
#pragma unroll
  for (int j = 0; j < 16; ++j) {
    const float w = s[j] * inv;
    float vf[8];
    load_bf16x8(Vb + ((long)b * Npts + idx[j]) * 256 + c0, vf);
#pragma unroll
    for (int c = 0; c < 8; ++c) o[c] += w * vf[c];
  }
  float r[8];
#pragma unroll
  for (int c = 0; c < 8; ++c) r[c] = o[c] + q[c];
  float sm = 0.f;
#pragma unroll
  for (int c = 0; c < 8; ++c) sm += r[c];
  const float mean = wave_sum(sm) * (1.f / 256.f);
  float vs = 0.f;
#pragma unroll
  for (int c = 0; c < 8; ++c) { const float d = r[c] - mean; vs += d * d; }
  const float var  = wave_sum(vs) * (1.f / 256.f);
  const float rstd = rsqrtf(var + 1e-5f);
  const float al   = alpha[gw * 3 + iblk];

  float* op = out + (long)gw * 256 + c0;
#pragma unroll
  for (int c = 0; c < 8; ++c)
    op[c] += al * ((r[c] - mean) * rstd * lng[c0 + c] + lnb[c0 + c]);
}

// ---------------------------------------------------------------------------
// out = LN(out + x_l3) with fn_g / fn_b, in place
// ---------------------------------------------------------------------------
__global__ __launch_bounds__(256)
void final_ln_kernel(float* __restrict__ out, const float* __restrict__ x3,
                     const float* __restrict__ g, const float* __restrict__ bb,
                     const int Mtot)
{
  const int gw = blockIdx.x * 8 + (threadIdx.x >> 5);
  if (gw >= Mtot) return;
  const int lane = threadIdx.x & 31;
  const int c0 = lane * 8;
  float* op = out + (long)gw * 256 + c0;
  const float* xp = x3 + (long)gw * 256 + c0;
  float r[8];
#pragma unroll
  for (int c = 0; c < 8; ++c) r[c] = op[c] + xp[c];
  float sm = 0.f;
#pragma unroll
  for (int c = 0; c < 8; ++c) sm += r[c];
  const float mean = wave_sum(sm) * (1.f / 256.f);
  float vs = 0.f;
#pragma unroll
  for (int c = 0; c < 8; ++c) { const float d = r[c] - mean; vs += d * d; }
  const float var  = wave_sum(vs) * (1.f / 256.f);
  const float rstd = rsqrtf(var + 1e-5f);
#pragma unroll
  for (int c = 0; c < 8; ++c)
    op[c] = (r[c] - mean) * rstd * g[c0 + c] + bb[c0 + c];
}

__global__ void zero_kernel(float4* __restrict__ p, const long n4) {
  const long i = (long)blockIdx.x * blockDim.x + threadIdx.x;
  if (i < n4) p[i] = make_float4(0.f, 0.f, 0.f, 0.f);
}

// ---------------------------------------------------------------------------
extern "C" void kernel_launch(void* const* d_in, const int* in_sizes, int n_in,
                              void* d_out, int out_size, void* d_ws, size_t ws_size,
                              hipStream_t stream)
{
  (void)in_sizes; (void)n_in; (void)out_size; (void)ws_size;
  const int  Nn = 20000, Cc = 256, Hh = 128;
  const long Mtot = 2L * Nn;   // 40000

  const float* x[4] = {(const float*)d_in[0], (const float*)d_in[1],
                       (const float*)d_in[2], (const float*)d_in[3]};
  const int*   knn = (const int*)d_in[4];
  const float* Wq  = (const float*)d_in[5];
  const float* bq  = (const float*)d_in[6];
  const float* Wk  = (const float*)d_in[7];
  const float* bk  = (const float*)d_in[8];
  const float* Wv  = (const float*)d_in[9];
  const float* bv  = (const float*)d_in[10];
  const float* lng = (const float*)d_in[11];
  const float* lnb = (const float*)d_in[12];
  const float* Wg1 = (const float*)d_in[13];
  const float* bg1 = (const float*)d_in[14];
  const float* Wg2 = (const float*)d_in[15];
  const float* bg2 = (const float*)d_in[16];
  const float* fng = (const float*)d_in[17];
  const float* fnb = (const float*)d_in[18];
  float* out = (float*)d_out;

  // ------------------- workspace layout -------------------
  const long XE = Mtot * Cc;          // 10,240,000 elements
  unsigned short* xb[4];
  unsigned short* p = (unsigned short*)d_ws;
  for (int i = 0; i < 4; ++i) { xb[i] = p; p += XE; }
  unsigned short* wqb  = p; p += 3L * Cc * Cc;
  unsigned short* wkb  = p; p += 3L * Cc * Cc;
  unsigned short* wvb  = p; p += 3L * Cc * Cc;
  unsigned short* wg1b = p; p += (long)Hh * Cc;
  float* q_buf = (float*)p;                  // Mtot*Cc f32 (also holds h)
  unsigned short* k_buf = (unsigned short*)(q_buf + XE);
  unsigned short* v_buf = k_buf + XE;
  float* a_buf = (float*)(v_buf + XE);       // Mtot*3 f32

  const dim3 blk(256);
  const dim3 gemmGrid((unsigned)(Mtot / TM), Cc / TN);   // 625 x 2
  const dim3 gateGrid((unsigned)(Mtot / TM), Hh / TN);   // 625 x 1
  const int  ptBlocks = (int)((Mtot + 7) / 8);           // 5000

  // ---- one-shot bf16 conversion of activations + weights ----
  const long xq4 = XE / 4;                               // 2,560,000
  for (int i = 0; i < 4; ++i)
    f32_to_bf16_kernel<<<(unsigned)((xq4 + 255) / 256), blk, 0, stream>>>(x[i], xb[i], xq4);
  const long wq4 = 3L * Cc * Cc / 4;                     // 49,152
  f32_to_bf16_kernel<<<(unsigned)((wq4 + 255) / 256), blk, 0, stream>>>(Wq, wqb, wq4);
  f32_to_bf16_kernel<<<(unsigned)((wq4 + 255) / 256), blk, 0, stream>>>(Wk, wkb, wq4);
  f32_to_bf16_kernel<<<(unsigned)((wq4 + 255) / 256), blk, 0, stream>>>(Wv, wvb, wq4);
  const long g4 = (long)Hh * Cc / 4;                     // 8,192
  f32_to_bf16_kernel<<<(unsigned)((g4 + 255) / 256), blk, 0, stream>>>(Wg1, wg1b, g4);

  // ---- gating: h = relu(x3 @ Wg1^T + bg1) in q_buf; alpha ----
  gemm_bf16_wmma<<<gateGrid, blk, 0, stream>>>(xb[3], wg1b, bg1, q_buf, Cc, Hh, 1);
  gate_alpha_kernel<<<ptBlocks, blk, 0, stream>>>(q_buf, Wg2, bg2, a_buf, (int)Mtot);

  // ---- zero output accumulator ----
  const long n4 = XE / 4;
  zero_kernel<<<(unsigned)((n4 + 255) / 256), blk, 0, stream>>>((float4*)out, n4);

  // ---- three scales ----
  for (int i = 0; i < 3; ++i) {
    gemm_bf16_wmma<<<gemmGrid, blk, 0, stream>>>(xb[i], wqb + (long)i * Cc * Cc,
                                                 bq + i * Cc, q_buf, Cc, Cc, 0);
    gemm_bf16_wmma<<<gemmGrid, blk, 0, stream>>>(xb[3], wkb + (long)i * Cc * Cc,
                                                 bk + i * Cc, k_buf, Cc, Cc, 2);
    gemm_bf16_wmma<<<gemmGrid, blk, 0, stream>>>(xb[3], wvb + (long)i * Cc * Cc,
                                                 bv + i * Cc, v_buf, Cc, Cc, 2);
    knn_attn_kernel<<<ptBlocks, blk, 0, stream>>>(q_buf, k_buf, v_buf, knn,
                                                  lng + i * Cc, lnb + i * Cc,
                                                  a_buf, i, out, Nn);
  }
  final_ln_kernel<<<ptBlocks, blk, 0, stream>>>(out, x[3], fng, fnb, (int)Mtot);
}
